// MultiheadAttention_29016799052527
// MI455X (gfx1250) — matrix operand order repaired and checked by reading, not compiled, not run on hardware
//
#include <hip/hip_runtime.h>
#include <hip/hip_bf16.h>
#include <math.h>

// ---------------------------------------------------------------------------
// MI455X (gfx1250) fused graphormer multi-head attention, bf16 WMMA pipeline.
//   d_in: 0=q 1=k 2=v (4,1024,1024) f32; 3=mask (4,1024) bool(1B);
//         4=graphormer_dists (4,1024,1024,8) f32; 5=weights (4096,1024) f32;
//         6=biases (4096) f32; 7=lin_w (16,8) f32
//   d_out: out (4,1024,1024) f32  ++  att (4,16,1024,1024) f32
//   d_ws : 8 x 8MB bf16 scratch buffers (64 MB)
// ---------------------------------------------------------------------------

typedef __bf16 bf16_t;
typedef __attribute__((ext_vector_type(16))) __bf16 v16bf;
typedef __attribute__((ext_vector_type(8)))  __bf16 v8bf;   // 16 bytes
typedef __attribute__((ext_vector_type(8)))  float  v8f;
typedef __attribute__((ext_vector_type(4)))  float  v4f;    // raw vector (NT-able)
typedef __attribute__((ext_vector_type(4)))  int    v4i;

union ABfrag { v16bf v; v8bf h[2]; };

// ---- CDNA5 feature gates (compile-safe on both toolchains) ----------------
#if __has_builtin(__builtin_amdgcn_global_load_async_to_lds_b128) && \
    __has_builtin(__builtin_amdgcn_s_wait_asynccnt)
#define HAVE_ASYNC_LDS 1
typedef __attribute__((address_space(1))) v4i gv4i;   // global-AS 16B chunk
typedef __attribute__((address_space(3))) v4i lv4i;   // LDS-AS    16B chunk
#else
#define HAVE_ASYNC_LDS 0
#endif

#if __has_builtin(__builtin_nontemporal_store) && __has_builtin(__builtin_nontemporal_load)
#define NT_STORE(v, p) __builtin_nontemporal_store((v), (p))
#define NT_LOAD(p)     __builtin_nontemporal_load((p))
#else
#define NT_STORE(v, p) (*(p) = (v))
#define NT_LOAD(p)     (*(p))
#endif

__device__ __forceinline__ bf16_t f2bf(float f) {
  unsigned u = __builtin_bit_cast(unsigned, f);
  u += 0x7FFFu + ((u >> 16) & 1u);               // round-to-nearest-even
  return __builtin_bit_cast(bf16_t, (unsigned short)(u >> 16));
}

// ------------------------------ f32 -> bf16 --------------------------------
__global__ __launch_bounds__(256) void cvt_f32_to_bf16(const float* __restrict__ src,
                                                       bf16_t* __restrict__ dst, int n) {
  int i = (blockIdx.x * 256 + threadIdx.x) * 4;
  if (i + 3 < n) {
    v4f f = *(const v4f*)(src + i);
    dst[i + 0] = f2bf(f.x);
    dst[i + 1] = f2bf(f.y);
    dst[i + 2] = f2bf(f.z);
    dst[i + 3] = f2bf(f.w);
  }
}

// ------------------------------ GEMM (X @ W^T + b) -------------------------
// X: M x 1024 bf16 row-major.  W: rows of the bf16 weight block, W[o][k].
// Block = 8 waves = 16 rows x 128 cols; each wave owns one 16x16 D tile.
// The 16x1024 A row-block (32 KB) is staged into LDS once per block via the
// gfx1250 async global->LDS path, then all waves read A fragments with ds ops.
// OUT_MODE 0: bf16 row-major   (Q/K projections)
// OUT_MODE 1: bf16 transposed per batch of 1024 rows (V -> VpT[b][chan][tok])
// OUT_MODE 2: f32 row-major    (final output projection -> d_out)
template <int OUT_MODE>
__global__ __launch_bounds__(256) void gemm16(const bf16_t* __restrict__ X,
                                              const bf16_t* __restrict__ W,
                                              const float* __restrict__ bias,
                                              void* __restrict__ outp) {
  const int K = 1024;
  __shared__ bf16_t Alds[16 * 1024];               // 32 KB staged A row-block

  const int wave = threadIdx.x >> 5, lane = threadIdx.x & 31;
  const int g = lane >> 4, r = lane & 15;
  const int m0 = blockIdx.x << 4;
  const int n0 = (blockIdx.y << 7) + (wave << 4);

  // ---- cooperative A stage: 2048 x 16B chunks, 8 per thread ---------------
  {
    const bf16_t* src = X + (size_t)m0 * K;        // 16 consecutive rows
#if HAVE_ASYNC_LDS
#pragma unroll
    for (int i = 0; i < 8; ++i) {
      const int c = threadIdx.x + i * 256;         // 16-byte chunk id
      __builtin_amdgcn_global_load_async_to_lds_b128(
          (gv4i*)(src + c * 8), (lv4i*)(Alds + c * 8), 0, 0);
    }
    __builtin_amdgcn_s_wait_asynccnt(0);
#else
#pragma unroll
    for (int i = 0; i < 8; ++i) {
      const int c = threadIdx.x + i * 256;
      *(v8bf*)(Alds + c * 8) = *(const v8bf*)(src + c * 8);
    }
#endif
    __syncthreads();
  }

  const bf16_t* arow = Alds + r * K;               // A row = lane%16 (LDS)
  const bf16_t* brow = W + (size_t)(n0 + r) * K;   // B col = lane%16 (global)
  v8f acc = {};

#pragma unroll 4
  for (int k0 = 0; k0 < K; k0 += 32) {
    if (k0 + 256 < K)
      __builtin_prefetch(brow + k0 + 256, 0, 0);   // global_prefetch_b8
    ABfrag a, b;
    // A layout: half h <-> K = (h>>3)*16 + g*8 + (h&7)  (ds_load_b128 x2)
    a.h[0] = *(const v8bf*)(arow + k0 + g * 8);
    a.h[1] = *(const v8bf*)(arow + k0 + 16 + g * 8);
    // B layout: half h <-> K = (h>>3)*16 + g*8 + (h&7)  (global_load_b128 x2)
    b.h[0] = *(const v8bf*)(brow + k0 + g * 8);
    b.h[1] = *(const v8bf*)(brow + k0 + 16 + g * 8);
    acc = __builtin_amdgcn_wmma_f32_16x16x32_bf16(false, a.v, false, b.v,
                                                  (short)0, acc, false, false);
  }
  const float bv = bias[n0 + r];

  if constexpr (OUT_MODE == 0) {
    bf16_t* out = (bf16_t*)outp;
#pragma unroll
    for (int v = 0; v < 8; ++v)
      out[(size_t)(m0 + g * 8 + v) * 1024 + n0 + r] = f2bf(acc[v] + bv);
  } else if constexpr (OUT_MODE == 1) {
    bf16_t* out = (bf16_t*)outp;
#pragma unroll
    for (int v = 0; v < 8; ++v) {
      int i = m0 + g * 8 + v;                       // global token row
      out[((size_t)(i >> 10) << 20) + (size_t)(n0 + r) * 1024 + (i & 1023)] =
          f2bf(acc[v] + bv);
    }
  } else {
    float* out = (float*)outp;
#pragma unroll
    for (int v = 0; v < 8; ++v)
      out[(size_t)(m0 + g * 8 + v) * 1024 + n0 + r] = acc[v] + bv;
  }
}

// --------------------------- fused attention -------------------------------
// One workgroup per (batch, head, 16-query tile).  8 waves / 256 threads.
// Phase 1: S[16][1024] = scale*Q.Kt + graphormer bias (+mask) via WMMA -> LDS
// Phase 2: exact softmax over the 1024 keys in LDS; att written NT to d_out
// Phase 3: O = P @ V via WMMA (A rebuilt from LDS probs), ctx written as bf16
#define SROW 1032  // 1024 + 8 floats pad: keeps 16B alignment, dodges bank camp

__global__ __launch_bounds__(256) void attn_fused(
    const bf16_t* __restrict__ Qp, const bf16_t* __restrict__ Kp,
    const bf16_t* __restrict__ VpT, const float* __restrict__ dists,
    const float* __restrict__ lin_w, const unsigned char* __restrict__ mask,
    float* __restrict__ att_out, bf16_t* __restrict__ ctx) {
  __shared__ float S[16 * SROW];        // 66 KB score/prob tile
  __shared__ float red[2][16][16];      // softmax reductions
  __shared__ float oxc[4][256];         // PV cross-wave reduction

  const int b = blockIdx.z, h = blockIdx.y, q0 = blockIdx.x << 4;
  const int wave = threadIdx.x >> 5, lane = threadIdx.x & 31;
  const int g = lane >> 4, r = lane & 15;
  const float scale = 0.125f;  // 1/sqrt(64)

  float lw[8];
#pragma unroll
  for (int p = 0; p < 8; ++p) lw[p] = lin_w[h * 8 + p];

  // Q A-fragments for head-dim chunks [0,32) and [32,64) -- reused by all key tiles
  const bf16_t* qrow = Qp + (size_t)(b * 1024 + q0 + r) * 1024 + h * 64;
  ABfrag qa[2];
#pragma unroll
  for (int c2 = 0; c2 < 2; ++c2) {
    qa[c2].h[0] = *(const v8bf*)(qrow + c2 * 32 + g * 8);
    qa[c2].h[1] = *(const v8bf*)(qrow + c2 * 32 + 16 + g * 8);
  }
  bool mrow[8];
#pragma unroll
  for (int v = 0; v < 8; ++v) mrow[v] = mask[b * 1024 + q0 + g * 8 + v] != 0;

  // ---- phase 1: scores ----------------------------------------------------
  for (int kt = wave; kt < 64; kt += 8) {
    const int n0 = kt << 4;
    const bf16_t* krow = Kp + (size_t)(b * 1024 + n0 + r) * 1024 + h * 64;
    v8f acc = {};
#pragma unroll
    for (int c2 = 0; c2 < 2; ++c2) {
      ABfrag kb;
      kb.h[0] = *(const v8bf*)(krow + c2 * 32 + g * 8);
      kb.h[1] = *(const v8bf*)(krow + c2 * 32 + 16 + g * 8);
      acc = __builtin_amdgcn_wmma_f32_16x16x32_bf16(false, qa[c2].v, false, kb.v,
                                                    (short)0, acc, false, false);
    }
    // D element v -> (query m = g*8+v, key = n0+r); fold graphormer bias + mask.
    // Masked rows skip the dists read entirely (saves ~half of 128 MB stream).
    const float* dbase = dists + ((size_t)(b * 1024 + q0) * 1024 + n0 + r) * 8;
#pragma unroll
    for (int v = 0; v < 8; ++v) {
      const int m = g * 8 + v;
      float sc;
      if (mrow[v]) {
        sc = -1e9f;  // masked query row -> softmax collapses to uniform 1/1024
      } else {
        const float* dp = dbase + (size_t)m * (1024 * 8);
        v4f d0 = NT_LOAD((const v4f*)dp);       // read-once stream: NT
        v4f d1 = NT_LOAD((const v4f*)(dp + 4));
        float gb = d0.x * lw[0];
        gb = fmaf(d0.y, lw[1], gb); gb = fmaf(d0.z, lw[2], gb);
        gb = fmaf(d0.w, lw[3], gb); gb = fmaf(d1.x, lw[4], gb);
        gb = fmaf(d1.y, lw[5], gb); gb = fmaf(d1.z, lw[6], gb);
        gb = fmaf(d1.w, lw[7], gb);
        sc = fmaf(acc[v], scale, gb);
      }
      S[m * SROW + n0 + r] = sc;
    }
  }
  __syncthreads();

  // ---- phase 2: softmax + att output --------------------------------------
  {
    const int row = threadIdx.x >> 4, ch = threadIdx.x & 15;  // 16 thr / row
    float* srow = S + row * SROW + ch * 64;
    float mx = -3.0e38f;
#pragma unroll 8
    for (int j = 0; j < 64; ++j) mx = fmaxf(mx, srow[j]);
    red[0][row][ch] = mx;
    __syncthreads();
    float rmax = red[0][row][0];
#pragma unroll
    for (int c = 1; c < 16; ++c) rmax = fmaxf(rmax, red[0][row][c]);
    float sum = 0.f;
#pragma unroll 8
    for (int j = 0; j < 64; ++j) {
      float e = __expf(srow[j] - rmax);
      srow[j] = e;
      sum += e;
    }
    red[1][row][ch] = sum;
    __syncthreads();
    float tot = 0.f;
#pragma unroll
    for (int c = 0; c < 16; ++c) tot += red[1][row][c];
    const float inv = 1.0f / tot;
    float* gout = att_out + ((size_t)((b * 16 + h) * 1024 + q0 + row) << 10) + ch * 64;
#pragma unroll 4
    for (int j = 0; j < 64; j += 4) {
      v4f pv;
      pv.x = srow[j + 0] * inv; pv.y = srow[j + 1] * inv;
      pv.z = srow[j + 2] * inv; pv.w = srow[j + 3] * inv;
      srow[j + 0] = pv.x; srow[j + 1] = pv.y;
      srow[j + 2] = pv.z; srow[j + 3] = pv.w;
      NT_STORE(pv, (v4f*)(gout + j));   // 256 MB write-once stream: NT
    }
  }
  __syncthreads();

  // ---- phase 3: O = P @ V -------------------------------------------------
  {
    const int ct = wave & 3;   // which 16-wide head-dim column tile
    const int kh = wave >> 2;  // which 512-key half
    const bf16_t* vcol = VpT + ((size_t)b << 20) + (size_t)(h * 64 + ct * 16 + r) * 1024;
    v8f oa = {};
    for (int k0 = kh * 512; k0 < kh * 512 + 512; k0 += 32) {
      ABfrag pa, vb;
      const float* sp = S + r * SROW + k0 + g * 8;  // probs, A row = query r
#pragma unroll
      for (int c2 = 0; c2 < 2; ++c2)
#pragma unroll
        for (int jj = 0; jj < 8; ++jj)
          pa.v[c2 * 8 + jj] = f2bf(sp[c2 * 16 + jj]);
      vb.h[0] = *(const v8bf*)(vcol + k0 + g * 8);
      vb.h[1] = *(const v8bf*)(vcol + k0 + 16 + g * 8);
      oa = __builtin_amdgcn_wmma_f32_16x16x32_bf16(false, pa.v, false, vb.v,
                                                   (short)0, oa, false, false);
    }
    if (kh == 1) {
#pragma unroll
      for (int v = 0; v < 8; ++v) oxc[ct][v * 32 + lane] = oa[v];
    }
    __syncthreads();
    if (kh == 0) {
#pragma unroll
      for (int v = 0; v < 8; ++v) {
        float o = oa[v] + oxc[ct][v * 32 + lane];
        // ctx[b*1024+tok][h*64 + d]  (matches transpose(0,2,1,3).reshape)
        ctx[(size_t)(b * 1024 + q0 + g * 8 + v) * 1024 + h * 64 + ct * 16 + r] =
            f2bf(o);
      }
    }
  }
}

// ---------------------------------------------------------------------------
extern "C" void kernel_launch(void* const* d_in, const int* in_sizes, int n_in,
                              void* d_out, int out_size, void* d_ws, size_t ws_size,
                              hipStream_t stream) {
  (void)in_sizes; (void)n_in; (void)out_size; (void)ws_size;

  const float*         q       = (const float*)d_in[0];
  const float*         k       = (const float*)d_in[1];
  const float*         v       = (const float*)d_in[2];
  const unsigned char* mask    = (const unsigned char*)d_in[3];  // bool[4096]
  const float*         dists   = (const float*)d_in[4];
  const float*         weights = (const float*)d_in[5];
  const float*         biases  = (const float*)d_in[6];
  const float*         lin_w   = (const float*)d_in[7];

  const size_t MN = 4ull * 1024 * 1024;  // 4096 x 1024 elements
  bf16_t* Wb  = (bf16_t*)d_ws;           // all 4 weight blocks, bf16
  bf16_t* Qb  = Wb  + MN;
  bf16_t* Kb  = Qb  + MN;
  bf16_t* Vb  = Kb  + MN;
  bf16_t* Qp  = Vb  + MN;                // projected Q (token-major)
  bf16_t* Kp  = Qp  + MN;                // projected K (token-major)
  bf16_t* VpT = Kp  + MN;                // projected V, transposed per batch
  bf16_t* Ctx = VpT + MN;                // attention context, pre-O-proj

  float* out_f = (float*)d_out;          // (4,1024,1024)
  float* att_f = out_f + MN;             // (4,16,1024,1024)

  // 1) f32 -> bf16 conversions
  const dim3 cblk(256), cgrd((unsigned)(MN / 1024));
  cvt_f32_to_bf16<<<cgrd, cblk, 0, stream>>>(weights, Wb, (int)MN);
  cvt_f32_to_bf16<<<cgrd, cblk, 0, stream>>>(q, Qb, (int)MN);
  cvt_f32_to_bf16<<<cgrd, cblk, 0, stream>>>(k, Kb, (int)MN);
  cvt_f32_to_bf16<<<cgrd, cblk, 0, stream>>>(v, Vb, (int)MN);

  // 2) QKV projections (bf16 WMMA, f32 accumulate)
  const dim3 gblk(256), ggrd(256, 8);
  gemm16<0><<<ggrd, gblk, 0, stream>>>(Qb, Wb,                   biases,        Qp);
  gemm16<0><<<ggrd, gblk, 0, stream>>>(Kb, Wb + 1024ull * 1024,  biases + 1024, Kp);
  gemm16<1><<<ggrd, gblk, 0, stream>>>(Vb, Wb + 2048ull * 1024,  biases + 2048, VpT);

  // 3) fused scores + bias + mask + softmax + PV
  attn_fused<<<dim3(64, 16, 4), 256, 0, stream>>>(Qp, Kp, VpT, dists, lin_w,
                                                  mask, att_f, Ctx);

  // 4) output projection -> f32 d_out
  gemm16<2><<<ggrd, gblk, 0, stream>>>(Ctx, Wb + 3072ull * 1024, biases + 3072, out_f);
}
